// paralle_attention2_61323543052446
// MI455X (gfx1250) — compile-verified, matrix operand order checked by reading, not run on hardware
//
#include <hip/hip_runtime.h>
#include <hip/hip_bf16.h>

// Problem dims (match reference)
#define B_   64
#define T_   256
#define D_   512
#define HA_  512
#define G4_  2048   // 4*HA
#define ODE_ 256
#define NBLK_ 128   // persistent scan: 128 blocks x 16 gate columns = 2048

typedef __bf16 bf16;
typedef __attribute__((ext_vector_type(16))) bf16  v16bf;
typedef __attribute__((ext_vector_type(8)))  float v8f;

__device__ __forceinline__ bf16 tobf(float f) {
  union { float f; unsigned u; } x; x.f = f;
  unsigned r = x.u + 0x7FFFu + ((x.u >> 16) & 1u);   // round-to-nearest-even
  union { unsigned short s; bf16 b; } y; y.s = (unsigned short)(r >> 16);
  return y.b;
}
__device__ __forceinline__ float frombf(bf16 b) {
  union { unsigned short s; bf16 b; } x; x.b = b;
  union { unsigned u; float f; } y; y.u = ((unsigned)x.s) << 16;
  return y.f;
}
__device__ __forceinline__ float sigmoidf_(float x) { return 1.0f / (1.0f + __expf(-x)); }

union FragU { v16bf v; uint4 q[2]; };

// WMMA 16x32 bf16 operand fragment loader (wave32).
// lane%16 = row; lanes 0-15 hold K = k0+{0..7,16..23}, lanes 16-31 K = k0+{8..15,24..31}.
__device__ __forceinline__ v16bf load_frag(const bf16* __restrict__ base, long ld,
                                           int lane, int k0) {
  int m  = lane & 15;
  int kb = k0 + ((lane >> 4) << 3);
  const bf16* p = base + (long)m * ld + kb;
  FragU f;
  f.q[0] = *reinterpret_cast<const uint4*>(p);
  f.q[1] = *reinterpret_cast<const uint4*>(p + 16);
  return f.v;
}

// One wave: 16(M) x 64(N) tile of A(16xK) * W(NxK)^T accumulated into acc[4].
__device__ __forceinline__ void wave_mma_16x64(const bf16* __restrict__ Abase, long lda,
                                               const bf16* __restrict__ Wbase, long ldw,
                                               int K, int lane, v8f acc[4]) {
  for (int k0 = 0; k0 < K; k0 += 32) {
    v16bf a = load_frag(Abase, lda, lane, k0);
#pragma unroll
    for (int nt = 0; nt < 4; ++nt) {
      v16bf b = load_frag(Wbase + (long)(nt * 16) * ldw, ldw, lane, k0);
      acc[nt] = __builtin_amdgcn_wmma_f32_16x16x32_bf16(
          false, a, false, b, (short)0, acc[nt], false, false);
    }
  }
}

// CDNA5 async copy: 16 bytes global -> LDS per enabled lane (ASYNCcnt-tracked).
__device__ __forceinline__ void async_copy_b128(unsigned lds_byte_off, const void* gptr) {
  asm volatile("global_load_async_to_lds_b128 %0, %1, off"
               :: "v"(lds_byte_off), "v"(gptr) : "memory");
}
__device__ __forceinline__ void wait_async0() {
  asm volatile("s_wait_asynccnt 0x0" ::: "memory");
}

// Device-wide sense-style barrier for the persistent scan kernel.
__device__ __forceinline__ void grid_barrier(unsigned* cnt, unsigned* gen, unsigned nb) {
  __threadfence();
  __syncthreads();
  if (threadIdx.x == 0) {
    unsigned g = __hip_atomic_load(gen, __ATOMIC_ACQUIRE, __HIP_MEMORY_SCOPE_AGENT);
    unsigned a = __hip_atomic_fetch_add(cnt, 1u, __ATOMIC_ACQ_REL, __HIP_MEMORY_SCOPE_AGENT);
    if (a + 1u == nb) {
      __hip_atomic_store(cnt, 0u, __ATOMIC_RELAXED, __HIP_MEMORY_SCOPE_AGENT);
      __hip_atomic_fetch_add(gen, 1u, __ATOMIC_RELEASE, __HIP_MEMORY_SCOPE_AGENT);
    } else {
      while (__hip_atomic_load(gen, __ATOMIC_ACQUIRE, __HIP_MEMORY_SCOPE_AGENT) == g)
        __builtin_amdgcn_s_sleep(2);
    }
  }
  __syncthreads();
  __threadfence();
}

// ---------------- prologue kernels ----------------
__global__ __launch_bounds__(256) void cvt_bf16_kernel(const float* __restrict__ s,
                                                       bf16* __restrict__ d, int n) {
  int i = blockIdx.x * 256 + threadIdx.x;
  if (i < n) d[i] = tobf(s[i]);
}

__global__ __launch_bounds__(256) void prep_kernel(const float* __restrict__ bih,
                                                   const float* __restrict__ bhh,
                                                   float* __restrict__ bias_a,
                                                   const float* __restrict__ ah0,
                                                   const float* __restrict__ ac0,
                                                   float* __restrict__ c,
                                                   bf16* __restrict__ hb,
                                                   unsigned* __restrict__ sync) {
  int i = blockIdx.x * 256 + threadIdx.x;
  if (i < G4_) bias_a[i] = bih[i] + bhh[i];
  if (i < B_ * HA_) { c[i] = ac0[i]; hb[i] = tobf(ah0[i]); }
  if (i < 2) sync[i] = 0u;   // grid-barrier counter + generation
}

// ---------------- emb = x @ W_emb^T + b_emb  (store bf16) ----------------
__global__ __launch_bounds__(256) void emb_gemm_kernel(const bf16* __restrict__ xb,
                                                       const bf16* __restrict__ Wemb,
                                                       const float* __restrict__ bemb,
                                                       bf16* __restrict__ embb) {
  int lane = threadIdx.x & 31, wave = threadIdx.x >> 5;
  int m0 = blockIdx.x * 128 + wave * 16;   // M = B*T = 16384
  int n0 = blockIdx.y * 64;                // N = 512
  v8f acc[4] = {};
  wave_mma_16x64(xb + (long)m0 * D_, D_, Wemb + (long)n0 * D_, D_, D_, lane, acc);
  int nl = lane & 15, mb = (lane >> 4) << 3;
#pragma unroll
  for (int nt = 0; nt < 4; ++nt) {
    int n = n0 + nt * 16 + nl;
    float bv = bemb[n];
#pragma unroll
    for (int r = 0; r < 8; ++r)
      embb[(long)(m0 + mb + r) * D_ + n] = tobf(acc[nt][r] + bv);
  }
}

// ---------------- persistent reverse-time LSTM scan ----------------
// 128 blocks x 128 threads (4 waves). Block j owns gate columns [j*16, j*16+16).
// Wih/Whh column slices (32KB) are staged into LDS ONCE via async-to-LDS, then
// all 256 steps run WMMA with B-fragments from LDS and A-fragments from L2.
__global__ __launch_bounds__(128) void lstm_scan_kernel(const bf16* __restrict__ embb,
                                                        const bf16* __restrict__ Wih,
                                                        const bf16* __restrict__ Whh,
                                                        bf16* __restrict__ hb,
                                                        float* __restrict__ c,
                                                        const float* __restrict__ bias,
                                                        const float* __restrict__ Waatt,
                                                        const float* __restrict__ baatt,
                                                        float* __restrict__ e,
                                                        float* __restrict__ G,
                                                        unsigned* __restrict__ sync) {
  __shared__ bf16 wsm[2 * 16 * D_];   // [0]=Wih slice, [1]=Whh slice (16 rows x 512)
  __shared__ float red[128];
  int lane = threadIdx.x & 31, wave = threadIdx.x >> 5;
  int n0 = blockIdx.x * 16;

  // -- one-time weight staging: 32KB per block, 16B chunks, ASYNCcnt-tracked --
  unsigned lds_base = (unsigned)(unsigned long long)(void*)wsm;  // low 32b = LDS offset
  for (int cch = threadIdx.x; cch < 2048; cch += 128) {
    int half = cch >> 10;            // 1024 chunks per slice
    int cc   = cch & 1023;
    int row  = cc >> 6;              // 64 x 16B chunks per 512-elem row
    int koff = (cc & 63) << 3;
    const bf16* src = (half ? Whh : Wih) + (long)(n0 + row) * D_ + koff;
    async_copy_b128(lds_base + (unsigned)(cch << 4), (const void*)src);
  }
  wait_async0();
  __syncthreads();

  unsigned* cnt = sync;
  unsigned* gen = sync + 1;
  const bf16* wih_s = wsm;
  const bf16* whh_s = wsm + 16 * D_;

  for (int s = 0; s < T_; ++s) {
    int t = T_ - 1 - s;

    // ---- gate GEMM phase: G[:, n0:n0+16] = emb_t@Wih^T + h@Whh^T + bias ----
    {
      int m0 = wave * 16;            // 4 waves cover B = 64 rows
      v8f acc = {};
      const bf16* Ax = embb + ((long)m0 * T_ + t) * D_;   // row stride T_*D_
      const bf16* Ah = hb + (long)m0 * HA_;
      for (int k0 = 0; k0 < D_; k0 += 32) {
        v16bf a = load_frag(Ax, (long)T_ * D_, lane, k0);
        v16bf b = load_frag(wih_s, D_, lane, k0);          // ds_load_b128 x2
        acc = __builtin_amdgcn_wmma_f32_16x16x32_bf16(
            false, a, false, b, (short)0, acc, false, false);
      }
      for (int k0 = 0; k0 < HA_; k0 += 32) {
        v16bf a = load_frag(Ah, HA_, lane, k0);
        v16bf b = load_frag(whh_s, HA_, lane, k0);
        acc = __builtin_amdgcn_wmma_f32_16x16x32_bf16(
            false, a, false, b, (short)0, acc, false, false);
      }
      int nl = lane & 15, mb = (lane >> 4) << 3;
      int n = n0 + nl;
      float bv = bias[n];
#pragma unroll
      for (int r = 0; r < 8; ++r)
        G[(long)(m0 + mb + r) * G4_ + n] = acc[r] + bv;
    }

    grid_barrier(cnt, gen, NBLK_);

    // ---- cell update + fused attention score (blocks 0..63, one batch each) ----
    if (blockIdx.x < B_) {
      int b = blockIdx.x;
      const float* Gr = G + (long)b * G4_;
      float part = 0.f;
      for (int d = threadIdx.x; d < HA_; d += 128) {
        float gi = Gr[d];
        float gf = Gr[HA_ + d];
        float gg = Gr[2 * HA_ + d];
        float go = Gr[3 * HA_ + d];
        float cn = sigmoidf_(gf) * c[b * HA_ + d] + sigmoidf_(gi) * tanhf(gg);
        float hn = sigmoidf_(go) * tanhf(cn);
        c[b * HA_ + d] = cn;
        hb[b * HA_ + d] = tobf(hn);
        part += hn * Waatt[d];
      }
      red[threadIdx.x] = part; __syncthreads();
      for (int s2 = 64; s2 > 0; s2 >>= 1) {
        if (threadIdx.x < s2) red[threadIdx.x] += red[threadIdx.x + s2];
        __syncthreads();
      }
      if (threadIdx.x == 0) e[b * T_ + t] = red[0] + baatt[0];
    }

    grid_barrier(cnt, gen, NBLK_);
  }
}

// ---------------- softmax over T per batch ----------------
__global__ __launch_bounds__(256) void softmax_kernel(const float* __restrict__ e,
                                                      float* __restrict__ alpha) {
  __shared__ float red[256];
  int b = blockIdx.x, t = threadIdx.x;
  float v = e[b * T_ + t];
  red[t] = v; __syncthreads();
  for (int s = 128; s > 0; s >>= 1) {
    if (t < s) red[t] = fmaxf(red[t], red[t + s]);
    __syncthreads();
  }
  float m = red[0]; __syncthreads();
  float ex = __expf(v - m);
  red[t] = ex; __syncthreads();
  for (int s = 128; s > 0; s >>= 1) {
    if (t < s) red[t] += red[t + s];
    __syncthreads();
  }
  alpha[b * T_ + t] = ex / red[0];
}

// ---------------- beta = tanh(hT @ W_batt^T + b_batt)  (time-invariant) ----------------
__global__ __launch_bounds__(128) void beta_gemm_kernel(const bf16* __restrict__ hb,
                                                        const bf16* __restrict__ Wbatt,
                                                        const float* __restrict__ bbatt,
                                                        float* __restrict__ beta) {
  int lane = threadIdx.x & 31, wave = threadIdx.x >> 5;
  int m0 = wave * 16;            // B = 64
  int n0 = blockIdx.x * 64;      // N = 512 -> 8 blocks
  v8f acc[4] = {};
  wave_mma_16x64(hb + (long)m0 * HA_, HA_, Wbatt + (long)n0 * HA_, HA_, HA_, lane, acc);
  int nl = lane & 15, mb = (lane >> 4) << 3;
#pragma unroll
  for (int nt = 0; nt < 4; ++nt) {
    int n = n0 + nt * 16 + nl;
    float bv = bbatt[n];
#pragma unroll
    for (int r = 0; r < 8; ++r)
      beta[(long)(m0 + mb + r) * D_ + n] = tanhf(acc[nt][r] + bv);
  }
}

// ---------------- var[b,d] = beta[b,d] * sum_t alpha[b,t]*emb[b,t,d] ----------------
__global__ __launch_bounds__(512) void weighted_reduce_kernel(const bf16* __restrict__ embb,
                                                              const float* __restrict__ alpha,
                                                              const float* __restrict__ beta,
                                                              float* __restrict__ var) {
  int b = blockIdx.x, d = threadIdx.x;   // 64 blocks x 512 threads
  const bf16* p = embb + (long)b * T_ * D_ + d;
  const float* a = alpha + b * T_;
  float s = 0.f;
  for (int t = 0; t < T_; ++t) s += a[t] * frombf(p[(long)t * D_]);
  var[b * D_ + d] = s * beta[b * D_ + d];
}

// ---------------- heads: mu = (var/T)@W_mu^T + b_mu ; sigma = var@W_sig^T + b_sig ----------------
__global__ __launch_bounds__(256) void heads_kernel(const float* __restrict__ var,
                                                    const float* __restrict__ Wmu,
                                                    const float* __restrict__ bmu,
                                                    const float* __restrict__ Wsig,
                                                    const float* __restrict__ bsig,
                                                    float* __restrict__ out) {
  int idx = blockIdx.x * 256 + threadIdx.x;
  if (idx >= B_ * ODE_) return;
  int b = idx / ODE_, o = idx % ODE_;
  const float* v  = var + b * D_;
  const float* wm = Wmu + (long)o * D_;
  const float* ws = Wsig + (long)o * D_;
  float sm = 0.f, ss = 0.f;
  for (int d = 0; d < D_; ++d) { float vv = v[d]; sm += vv * wm[d]; ss += vv * ws[d]; }
  out[idx]             = sm * (1.0f / (float)T_) + bmu[o];   // mu
  out[B_ * ODE_ + idx] = ss + bsig[o];                       // sigma
}

extern "C" void kernel_launch(void* const* d_in, const int* in_sizes, int n_in,
                              void* d_out, int out_size, void* d_ws, size_t ws_size,
                              hipStream_t stream) {
  const float* x      = (const float*)d_in[0];
  const float* W_emb  = (const float*)d_in[1];
  const float* b_emb  = (const float*)d_in[2];
  const float* Wih_a  = (const float*)d_in[3];
  const float* Whh_a  = (const float*)d_in[4];
  const float* bih_a  = (const float*)d_in[5];
  const float* bhh_a  = (const float*)d_in[6];
  // d_in[7..10]: beta-LSTM params -> dead code in reference, unused.
  const float* W_aatt = (const float*)d_in[11];
  const float* b_aatt = (const float*)d_in[12];
  const float* W_batt = (const float*)d_in[13];
  const float* b_batt = (const float*)d_in[14];
  const float* W_mu   = (const float*)d_in[15];
  const float* b_mu   = (const float*)d_in[16];
  const float* W_sig  = (const float*)d_in[17];
  const float* b_sig  = (const float*)d_in[18];
  const float* ah0    = (const float*)d_in[19];
  const float* ac0    = (const float*)d_in[20];
  float* out = (float*)d_out;

  // ---- workspace carve-up (256B aligned) ----
  size_t off = 0;
  auto carve = [&](size_t bytes) -> char* {
    char* p = (char*)d_ws + off;
    off += (bytes + 255) & ~(size_t)255;
    return p;
  };
  const long NX = (long)B_ * T_ * D_;      // 8,388,608
  bf16*     xb      = (bf16*)carve(NX * 2);
  bf16*     embb    = (bf16*)carve(NX * 2);
  bf16*     Wihb    = (bf16*)carve((long)G4_ * D_ * 2);
  bf16*     Whhb    = (bf16*)carve((long)G4_ * HA_ * 2);
  bf16*     Wembb   = (bf16*)carve((long)D_ * D_ * 2);
  bf16*     Wbattb  = (bf16*)carve((long)D_ * HA_ * 2);
  float*    bias_a  = (float*)carve(G4_ * 4);
  float*    G       = (float*)carve((long)B_ * G4_ * 4);
  float*    c_f     = (float*)carve((long)B_ * HA_ * 4);
  bf16*     hb      = (bf16*)carve((long)B_ * HA_ * 2);
  float*    e       = (float*)carve((long)B_ * T_ * 4);
  float*    alpha   = (float*)carve((long)B_ * T_ * 4);
  float*    beta    = (float*)carve((long)B_ * D_ * 4);
  float*    var     = (float*)carve((long)B_ * D_ * 4);
  unsigned* syncv   = (unsigned*)carve(2 * 4);
  (void)ws_size; (void)n_in; (void)in_sizes; (void)out_size;

  // ---- prologue: bf16 conversions, bias fold, state init, sync reset ----
  cvt_bf16_kernel<<<(int)((NX + 255) / 256), 256, 0, stream>>>(x, xb, (int)NX);
  cvt_bf16_kernel<<<(D_ * D_ + 255) / 256, 256, 0, stream>>>(W_emb, Wembb, D_ * D_);
  cvt_bf16_kernel<<<(G4_ * D_ + 255) / 256, 256, 0, stream>>>(Wih_a, Wihb, G4_ * D_);
  cvt_bf16_kernel<<<(G4_ * HA_ + 255) / 256, 256, 0, stream>>>(Whh_a, Whhb, G4_ * HA_);
  cvt_bf16_kernel<<<(D_ * HA_ + 255) / 256, 256, 0, stream>>>(W_batt, Wbattb, D_ * HA_);
  prep_kernel<<<(B_ * HA_ + 255) / 256, 256, 0, stream>>>(bih_a, bhh_a, bias_a,
                                                          ah0, ac0, c_f, hb, syncv);

  // ---- emb GEMM: 16384x512x512 ----
  emb_gemm_kernel<<<dim3((B_ * T_) / 128, D_ / 64), 256, 0, stream>>>(xb, Wembb, b_emb, embb);

  // ---- persistent reverse-time LSTM scan: ONE launch for all 256 steps ----
  lstm_scan_kernel<<<dim3(NBLK_), 128, 0, stream>>>(embb, Wihb, Whhb, hb, c_f, bias_a,
                                                    W_aatt, b_aatt, e, G, syncv);

  // ---- attention softmax, beta, reduction, heads ----
  softmax_kernel<<<dim3(B_), 256, 0, stream>>>(e, alpha);
  beta_gemm_kernel<<<dim3(D_ / 64), 128, 0, stream>>>(hb, Wbattb, b_batt, beta);
  weighted_reduce_kernel<<<dim3(B_), 512, 0, stream>>>(embb, alpha, beta, var);
  heads_kernel<<<(B_ * ODE_ + 255) / 256, 256, 0, stream>>>(var, W_mu, b_mu, W_sig, b_sig, out);
}
